// NN_C_61993557950503
// MI455X (gfx1250) — matrix operand order repaired and checked by reading, not compiled, not run on hardware
//
#include <hip/hip_runtime.h>

#define G_    32
#define N_    64
#define T_    20
#define DIM_  64
#define R_TOT (G_*N_*N_*T_)      // 2,621,440 rows
#define TILES (R_TOT/16)         // 163,840 16-row tiles
#define WPB   8                  // waves per block (256 threads, wave32)
#define TPW   16                 // tiles per wave (weights hoisted across them)
#define HS    72                 // f16 h row stride in halves (bank-skewed)

typedef __attribute__((ext_vector_type(2)))  float    v2f;
typedef __attribute__((ext_vector_type(8)))  float    v8f;
typedef __attribute__((ext_vector_type(8)))  _Float16 v8h;
typedef __attribute__((ext_vector_type(16))) _Float16 v16h;

// lrelu(x) == max(x, 0.1*x) for every x: 2 dual-issuable VALU ops, no cndmask
__device__ __forceinline__ float lrelu(float x) { return fmaxf(x, 0.1f * x); }

struct WaveLds {
    _Float16 h[16 * HS];   // 16x64 activations, f16, padded rows
    float    f[64];        // feat 16x4 (residual)
    float    y[64];        // y    16x4
    float    o[32];        // out staging 16x2
};

__global__ __launch_bounds__(256) void social_mask_mlp(
    const float* __restrict__ traj, const float* __restrict__ speed,
    const float* __restrict__ W1, const float* __restrict__ b1,
    const float* __restrict__ W2, const float* __restrict__ b2,
    const float* __restrict__ W3, const float* __restrict__ b3,
    float* __restrict__ mask_out, float* __restrict__ outs)
{
    __shared__ WaveLds lds[WPB];

    const int wave  = threadIdx.x >> 5;
    const int lane  = threadIdx.x & 31;
    const int col   = lane & 15;
    const bool hi   = lane >= 16;
    const int kbase = hi ? 2 : 0;

    _Float16* hL = lds[wave].h;
    float*    fL = lds[wave].f;
    float*    yL = lds[wave].y;
    float*    oL = lds[wave].o;

    // ---------------- tile-invariant fragments (built once per wave) --------
    // GEMM1: B tiles of W1 (4x16 each) + broadcast-bias C fragments
    v2f b1t[4];
    v8f c1[4];
    #pragma unroll
    for (int n = 0; n < 4; ++n) {
        b1t[n].x = W1[ kbase      * DIM_ + n*16 + col];
        b1t[n].y = W1[(kbase + 1) * DIM_ + n*16 + col];
        float bias = b1[n*16 + col];
        #pragma unroll
        for (int k = 0; k < 8; ++k) c1[n][k] = bias;
    }
    // GEMM2: two 32x16 f16 B fragments of W2 (cols 0..3 valid) + C bias
    // B k order chosen here (contiguous, A matches): element e of v16h <-> K = 32*c + 16*hi + e, N = col
    v16h b2f[2];
    #pragma unroll
    for (int c = 0; c < 2; ++c)
        #pragma unroll
        for (int e = 0; e < 16; ++e) {
            int kk = 32*c + (hi ? 16 : 0) + e;
            b2f[c][e] = (col < 4) ? (_Float16)W2[kk*4 + col] : (_Float16)0.0f;
        }
    float bias2 = (col < 4) ? b2[col] : 0.0f;
    v8f c2;
    #pragma unroll
    for (int k = 0; k < 8; ++k) c2[k] = bias2;
    // GEMM3: 4x16 B fragment of W3 (cols 0..1 valid) + C bias
    v2f b3t;
    b3t.x = (col < 2) ? W3[ kbase      * 2 + col] : 0.0f;
    b3t.y = (col < 2) ? W3[(kbase + 1) * 2 + col] : 0.0f;
    float bias3 = (col < 2) ? b3[col] : 0.0f;
    v8f c3;
    #pragma unroll
    for (int k = 0; k < 8; ++k) c3[k] = bias3;

    const int wave_id   = blockIdx.x * WPB + wave;
    const int tile_base = wave_id * TPW;

    for (int tt = 0; tt < TPW; ++tt) {
        const int tile    = tile_base + tt;
        const int rowbase = tile * 16;
        const int r       = rowbase + col;

        // r -> (g, i, j, t) for layout [G,N,N,T]
        unsigned t  = (unsigned)r % T_;
        unsigned q  = (unsigned)r / T_;
        unsigned j  = q  % N_;
        unsigned q2 = q  / N_;
        unsigned i  = q2 % N_;
        unsigned g  = q2 / N_;

        const float* pj = traj  + (((g*N_ + j)*T_ + t) << 1);
        const float* pi = traj  + (((g*N_ + i)*T_ + t) << 1);
        const float* vi = speed + (((g*N_ + i)*T_ + t) << 1);

        float pjx = pj[0], pjy = pj[1];
        float pix = pi[0], piy = pi[1];
        float vx  = vi[0], vy  = vi[1];
        float vecx = pjx - pix, vecy = pjy - piy;

        // mask: arccos removed (ang<=thr <=> cos-ratio>=cos(thr)); EPS makes
        // fast v_rcp/v_sqrt safe
        float dist = __builtin_amdgcn_sqrtf(vecx*vecx + vecy*vecy);
        float spd  = __builtin_amdgcn_sqrtf(vx*vx + vy*vy) + 1e-5f;
        float cr   = (vx*vecx + vy*vecy)
                   * __builtin_amdgcn_rcpf(dist + 1e-5f)
                   * __builtin_amdgcn_rcpf(spd);
        cr = fminf(fmaxf(cr, -1.0f), 1.0f);
        float m = (dist <= 2.0f && cr >= 0.2588696f) ? 1.0f : 0.0f;
        mask_out[r] = m;   // lanes L and L+16 write identical value -> branchless

        // A1 = feat fragment; the feat stash is the same data -> one b64 store
        v2f a1;
        a1.x = hi ? vx : vecx;
        a1.y = hi ? vy : vecy;
        *(v2f*)(fL + col*4 + kbase) = a1;

        // ---- GEMM1: h = lrelu(feat @ W1 + b1): 4 f32 K=4 WMMAs -> f16 LDS ----
        #pragma unroll
        for (int n = 0; n < 4; ++n) {
            v8f d = __builtin_amdgcn_wmma_f32_16x16x4_f32(
                        false, a1, false, b1t[n], (short)0, c1[n], false, false);
            #pragma unroll
            for (int k = 0; k < 8; ++k) {
                int row = k + (hi ? 8 : 0);
                hL[row*HS + n*16 + col] = (_Float16)lrelu(d[k]);
            }
        }
        asm volatile("s_wait_dscnt 0" ::: "memory");   // same-wave DS is in-order

        // ---- GEMM2: acc2 = h @ W2 + b2: two f16 K=32 WMMAs (chain from c2) ----
        const _Float16* base0 = hL + col*HS + (hi ? 16 : 0);
        v8h lo0 = *(const v8h*)(base0);
        v8h hi0 = *(const v8h*)(base0 + 8);
        v8h lo1 = *(const v8h*)(base0 + 32);
        v8h hi1 = *(const v8h*)(base0 + 40);
        v16h a2_0 = __builtin_shufflevector(lo0, hi0,
                      0,1,2,3,4,5,6,7,8,9,10,11,12,13,14,15);
        v16h a2_1 = __builtin_shufflevector(lo1, hi1,
                      0,1,2,3,4,5,6,7,8,9,10,11,12,13,14,15);
        v8f acc2 = __builtin_amdgcn_wmma_f32_16x16x32_f16(
                       false, a2_0, false, b2f[0], (short)0, c2, false, false);
        acc2 = __builtin_amdgcn_wmma_f32_16x16x32_f16(
                       false, a2_1, false, b2f[1], (short)0, acc2, false, false);

        // ---- y = lrelu(acc2) + feat (cols 0..3 valid) -> row-major LDS ----
        if (col < 4) {
            #pragma unroll
            for (int k = 0; k < 8; ++k) {
                int row = k + (hi ? 8 : 0);
                yL[row*4 + col] = lrelu(acc2[k]) + fL[row*4 + col];
            }
        }
        asm volatile("s_wait_dscnt 0" ::: "memory");

        // ---- GEMM3: out = sigmoid(y @ W3 + b3): one f32 K=4 WMMA ----
        v2f a3 = *(const v2f*)(yL + col*4 + kbase);
        v8f d3 = __builtin_amdgcn_wmma_f32_16x16x4_f32(
                     false, a3, false, b3t, (short)0, c3, false, false);

        if (col < 2) {
            #pragma unroll
            for (int k = 0; k < 8; ++k) {
                int row = k + (hi ? 8 : 0);
                oL[row*2 + col] = __builtin_amdgcn_rcpf(1.0f + __expf(-d3[k]));
            }
        }
        asm volatile("s_wait_dscnt 0" ::: "memory");

        // coalesced 128B store of the 16x2 output tile
        outs[rowbase*2 + lane] = oL[lane];
    }
}

extern "C" void kernel_launch(void* const* d_in, const int* in_sizes, int n_in,
                              void* d_out, int out_size, void* d_ws, size_t ws_size,
                              hipStream_t stream) {
    const float* traj  = (const float*)d_in[0];
    const float* speed = (const float*)d_in[1];
    // d_in[2] = seq_start_end (contiguous equal groups; index math replaces it)
    const float* W1 = (const float*)d_in[3];
    const float* b1 = (const float*)d_in[4];
    const float* W2 = (const float*)d_in[5];
    const float* b2 = (const float*)d_in[6];
    const float* W3 = (const float*)d_in[7];
    const float* b3 = (const float*)d_in[8];

    float* out      = (float*)d_out;
    float* mask_out = out;            // [G,N,N,T] as 0.0/1.0
    float* outs     = out + R_TOT;    // [G,N,N,T,2]

    (void)in_sizes; (void)n_in; (void)out_size; (void)d_ws; (void)ws_size;

    dim3 grid(TILES / (WPB * TPW));   // 1,280 blocks x 256 threads
    social_mask_mlp<<<grid, 256, 0, stream>>>(traj, speed, W1, b1, W2, b2, W3, b3,
                                              mask_out, outs);
}
